// QueryConditionedAttention_90460601189048
// MI455X (gfx1250) — matrix operand order repaired and checked by reading, not compiled, not run on hardware
//
#include <hip/hip_runtime.h>
#include <math.h>

#define HIDDEN 128
#define HEADS  8
#define HD     16

typedef __attribute__((ext_vector_type(16))) __bf16 v16bf;
typedef __attribute__((ext_vector_type(8)))  __bf16 v8bf;
typedef __attribute__((ext_vector_type(8)))  float  v8f;

// ---------------------------------------------------------------------------
// Utility kernels
// ---------------------------------------------------------------------------
__global__ void fill_zero_f32(float* __restrict__ p, long n) {
    long i = (long)blockIdx.x * blockDim.x + threadIdx.x;
    long stride = (long)gridDim.x * blockDim.x;
    for (; i < n; i += stride) p[i] = 0.0f;
}

__global__ void cvt_f32_to_bf16(const float* __restrict__ in, __bf16* __restrict__ out, long n) {
    long i = (long)blockIdx.x * blockDim.x + threadIdx.x;
    long stride = (long)gridDim.x * blockDim.x;
    for (; i < n; i += stride) out[i] = (__bf16)in[i];
}

// Transpose+convert the four 128x128 weight blocks into bf16 Wt[n][k]
// (Wq/Wk: only the top 128 rows multiply node_features; the query half is
//  folded into constant bias vectors cq/ck by prep_consts below.)
__global__ void prep_weights(const float* __restrict__ Wq, const float* __restrict__ Wk,
                             const float* __restrict__ Wv, const float* __restrict__ Wo,
                             __bf16* __restrict__ Wqt, __bf16* __restrict__ Wkt,
                             __bf16* __restrict__ Wvt, __bf16* __restrict__ Wot) {
    int idx = blockIdx.x * blockDim.x + threadIdx.x;   // 4*128*128
    if (idx >= 4 * HIDDEN * HIDDEN) return;
    int mat = idx >> 14;
    int rem = idx & (HIDDEN * HIDDEN - 1);
    int ncol = rem >> 7;
    int k    = rem & (HIDDEN - 1);
    const float* src; __bf16* dst;
    switch (mat) {
        case 0:  src = Wq; dst = Wqt; break;
        case 1:  src = Wk; dst = Wkt; break;
        case 2:  src = Wv; dst = Wvt; break;
        default: src = Wo; dst = Wot; break;
    }
    dst[ncol * HIDDEN + k] = (__bf16)src[k * HIDDEN + ncol];
}

// cq[j] = bq[j] + sum_i query[i] * Wq[128+i][j]   (query broadcast collapses)
__global__ void prep_consts(const float* __restrict__ qe,
                            const float* __restrict__ Wq, const float* __restrict__ bq,
                            const float* __restrict__ Wk, const float* __restrict__ bk,
                            float* __restrict__ cq, float* __restrict__ ck) {
    int j = threadIdx.x;                 // 128 threads
    float sq = bq[j], sk = bk[j];
    for (int i = 0; i < HIDDEN; ++i) {
        float q = qe[i];
        sq += q * Wq[(HIDDEN + i) * HIDDEN + j];
        sk += q * Wk[(HIDDEN + i) * HIDDEN + j];
    }
    cq[j] = sq;
    ck[j] = sk;
}

// ---------------------------------------------------------------------------
// bf16 WMMA GEMM: Out[M,128] = A[M,128] @ Bt^T + bias   (Bt is [n][k] bf16)
// One wave -> one 16-column tile; the wave's full B strip (16 cols x K=128,
// 4 x v16bf = 32 VGPRs) is loaded ONCE and reused across 4 row tiles per
// block.  8 waves/block cover all 128 output columns.
//
// A layout (16-bit A 16x32): lanes 0-15 & 16-31 both hold rows M=0..15,
//   half=lane>>4 selects K sub-ranges; per-lane elements are one contiguous
//   16-bf16 chunk at k0+half*16 (same k order as B).
// B layout (dense 32x16): lane -> N=lane&15, half selects K=0..15 / 16..31,
//   i.e. one contiguous 16-bf16 chunk of Bt row N at k0+half*16.
// C/D layout: VGPR r -> (M = r + 8*half, N = lane&15).
// ---------------------------------------------------------------------------
__device__ __forceinline__ v8f wmma_kstep(const __bf16* __restrict__ arow,
                                          int k0, int half, v16bf b, v8f acc) {
    v16bf a = *(const v16bf*)(arow + k0 + half * 16);
    return __builtin_amdgcn_wmma_f32_16x16x32_bf16(
               false, a, false, b, (short)0, acc, false, false);
}

#define ROWTILES_PER_BLOCK 4

__global__ void gemm128_bf16_wmma(const __bf16* __restrict__ A,
                                  const __bf16* __restrict__ Bt,
                                  const float*  __restrict__ bias,
                                  float* __restrict__ Out, long M) {
    const int lane = threadIdx.x & 31;
    const int wave = threadIdx.x >> 5;          // column tile 0..7
    const int half = lane >> 4;
    const int m    = lane & 15;
    const int colBase = wave * 16;

    // Hoisted B strip: columns colBase..colBase+15, K = 0..127.
    const __bf16* brow = Bt + (long)(colBase + m) * HIDDEN;
    const v16bf b0 = *(const v16bf*)(brow +  0 + half * 16);
    const v16bf b1 = *(const v16bf*)(brow + 32 + half * 16);
    const v16bf b2 = *(const v16bf*)(brow + 64 + half * 16);
    const v16bf b3 = *(const v16bf*)(brow + 96 + half * 16);
    const float bcol = bias[colBase + m];

    #pragma unroll
    for (int t = 0; t < ROWTILES_PER_BLOCK; ++t) {
        const long rowBase = ((long)blockIdx.x * ROWTILES_PER_BLOCK + t) * 16;
        if (rowBase >= M) break;                 // uniform branch

        long aRow = rowBase + m;
        if (aRow >= M) aRow = M - 1;             // clamp loads; EXEC stays full for WMMA
        const __bf16* arow = A + aRow * HIDDEN;

        v8f acc = {};
        acc = wmma_kstep(arow,  0, half, b0, acc);
        acc = wmma_kstep(arow, 32, half, b1, acc);
        acc = wmma_kstep(arow, 64, half, b2, acc);
        acc = wmma_kstep(arow, 96, half, b3, acc);

        if (rowBase + 16 <= M) {                 // full tile: branch-free stores
            float* op = Out + (rowBase + half * 8) * HIDDEN + colBase + m;
            #pragma unroll
            for (int r = 0; r < 8; ++r)
                op[(long)r * HIDDEN] = acc[r] + bcol;
        } else {                                 // tail tile (unused when M%16==0)
            #pragma unroll
            for (int r = 0; r < 8; ++r) {
                long oRow = rowBase + half * 8 + r;
                if (oRow < M)
                    Out[oRow * HIDDEN + colBase + m] = acc[r] + bcol;
            }
        }
    }
}

// ---------------------------------------------------------------------------
// Edge phase. One thread per (edge, head).
// ---------------------------------------------------------------------------
__global__ void edge_scores(const float* __restrict__ Q, const float* __restrict__ K,
                            const float* __restrict__ rel,
                            const int* __restrict__ ei, const int* __restrict__ et,
                            float* __restrict__ scores, float* __restrict__ segmax,
                            long E) {
    long idx = (long)blockIdx.x * blockDim.x + threadIdx.x;
    if (idx >= E * HEADS) return;
    long e = idx >> 3;
    int  h = (int)(idx & 7);
    int src = ei[e], dst = ei[E + e], r = et[e];

    const float4* qp = (const float4*)(Q   + (long)dst * HIDDEN + h * HD);
    const float4* kp = (const float4*)(K   + (long)src * HIDDEN + h * HD);
    const float4* rp = (const float4*)(rel + (long)r   * HIDDEN + h * HD);
    float s = 0.0f;
    #pragma unroll
    for (int i = 0; i < 4; ++i) {
        float4 qv = qp[i], kv = kp[i], rv = rp[i];
        s += qv.x * (kv.x + rv.x) + qv.y * (kv.y + rv.y)
           + qv.z * (kv.z + rv.z) + qv.w * (kv.w + rv.w);
    }
    s *= 0.25f;                                  // 1/sqrt(HEAD_DIM), HD=16
    scores[idx] = s;
    // segment max is clamped at 0 in the reference; buffer is init'd to 0, so
    // only positive scores can win -> positive-float-as-int max is exact.
    if (s > 0.0f)
        atomicMax((int*)(segmax + (long)dst * HEADS + h), __float_as_int(s));
}

__global__ void edge_exp(const int* __restrict__ ei,
                         float* __restrict__ scores,
                         const float* __restrict__ segmax,
                         float* __restrict__ segsum, long E) {
    long idx = (long)blockIdx.x * blockDim.x + threadIdx.x;
    if (idx >= E * HEADS) return;
    long e = idx >> 3;
    int  h = (int)(idx & 7);
    int dst = ei[E + e];
    float ex = __expf(scores[idx] - segmax[(long)dst * HEADS + h]);
    scores[idx] = ex;
    atomicAdd(segsum + (long)dst * HEADS + h, ex);
}

__global__ void edge_scatter(const int* __restrict__ ei,
                             const float* __restrict__ scores,
                             const float* __restrict__ segsum,
                             const float* __restrict__ V,
                             float* __restrict__ agg, long E) {
    long idx = (long)blockIdx.x * blockDim.x + threadIdx.x;
    if (idx >= E * HEADS) return;
    long e = idx >> 3;
    int  h = (int)(idx & 7);
    int src = ei[e], dst = ei[E + e];
    float attn = scores[idx] / (segsum[(long)dst * HEADS + h] + 1e-8f);
    const float4* vp = (const float4*)(V + (long)src * HIDDEN + h * HD);
    float* ap = agg + (long)dst * HIDDEN + h * HD;
    #pragma unroll
    for (int i = 0; i < 4; ++i) {
        float4 v = vp[i];
        atomicAdd(ap + 4 * i + 0, attn * v.x);
        atomicAdd(ap + 4 * i + 1, attn * v.y);
        atomicAdd(ap + 4 * i + 2, attn * v.z);
        atomicAdd(ap + 4 * i + 3, attn * v.w);
    }
}

// ---------------------------------------------------------------------------
// Host-side launch
// ---------------------------------------------------------------------------
extern "C" void kernel_launch(void* const* d_in, const int* in_sizes, int n_in,
                              void* d_out, int out_size, void* d_ws, size_t ws_size,
                              hipStream_t stream) {
    const float* nodef = (const float*)d_in[0];
    const float* qe    = (const float*)d_in[1];
    const float* rel   = (const float*)d_in[2];
    const float* Wq    = (const float*)d_in[3];
    const float* bq    = (const float*)d_in[4];
    const float* Wk    = (const float*)d_in[5];
    const float* bk    = (const float*)d_in[6];
    const float* Wv    = (const float*)d_in[7];
    const float* bv    = (const float*)d_in[8];
    const float* Wo    = (const float*)d_in[9];
    const float* bo    = (const float*)d_in[10];
    const int*   ei    = (const int*)d_in[11];
    const int*   et    = (const int*)d_in[12];

    const long N = in_sizes[0] / HIDDEN;
    const long E = in_sizes[12];

    // Workspace carve-out (256B aligned slices).
    char* base = (char*)d_ws;
    size_t off = 0;
    auto take = [&](size_t bytes) -> char* {
        char* p = base + off;
        off += (bytes + 255) & ~(size_t)255;
        return p;
    };
    float*  Qb     = (float*)take((size_t)N * HIDDEN * 4);
    float*  Kb     = (float*)take((size_t)N * HIDDEN * 4);
    float*  Vb     = (float*)take((size_t)N * HIDDEN * 4);
    float*  scores = (float*)take((size_t)E * HEADS * 4);
    float*  segmax = (float*)take((size_t)N * HEADS * 4);
    float*  segsum = (float*)take((size_t)N * HEADS * 4);
    float*  agg    = (float*)take((size_t)N * HIDDEN * 4);
    float*  cq     = (float*)take(HIDDEN * 4);
    float*  ck     = (float*)take(HIDDEN * 4);
    __bf16* Wqt    = (__bf16*)take(HIDDEN * HIDDEN * 2);
    __bf16* Wkt    = (__bf16*)take(HIDDEN * HIDDEN * 2);
    __bf16* Wvt    = (__bf16*)take(HIDDEN * HIDDEN * 2);
    __bf16* Wot    = (__bf16*)take(HIDDEN * HIDDEN * 2);
    __bf16* Xbf    = (__bf16*)take((size_t)N * HIDDEN * 2);  // node feats, later reused for agg

    const int BLK = 256;
    auto blocks = [](long work, int blk) { return (int)((work + blk - 1) / blk); };

    // 0) zero the accumulation buffers (harness poisons ws)
    fill_zero_f32<<<blocks(N * HEADS, BLK), BLK, 0, stream>>>(segmax, N * HEADS);
    fill_zero_f32<<<blocks(N * HEADS, BLK), BLK, 0, stream>>>(segsum, N * HEADS);
    fill_zero_f32<<<blocks(N * HIDDEN, BLK), BLK, 0, stream>>>(agg, N * HIDDEN);

    // 1) fold query contribution + bias into constant vectors; stage weights
    prep_consts<<<1, HIDDEN, 0, stream>>>(qe, Wq, bq, Wk, bk, cq, ck);
    prep_weights<<<blocks(4 * HIDDEN * HIDDEN, BLK), BLK, 0, stream>>>(
        Wq, Wk, Wv, Wo, Wqt, Wkt, Wvt, Wot);
    cvt_f32_to_bf16<<<blocks(N * HIDDEN, BLK), BLK, 0, stream>>>(nodef, Xbf, N * HIDDEN);

    // 2) Q/K/V projections via WMMA (each block covers 4 row tiles x 128 cols)
    const long rowTiles = (N + 15) / 16;
    const int  gemmBlocks = (int)((rowTiles + ROWTILES_PER_BLOCK - 1) / ROWTILES_PER_BLOCK);
    gemm128_bf16_wmma<<<gemmBlocks, BLK, 0, stream>>>(Xbf, Wqt, cq, Qb, N);
    gemm128_bf16_wmma<<<gemmBlocks, BLK, 0, stream>>>(Xbf, Wkt, ck, Kb, N);
    gemm128_bf16_wmma<<<gemmBlocks, BLK, 0, stream>>>(Xbf, Wvt, bv, Vb, N);

    // 3) edge phase: scores+segmax, exp+segsum, normalize+scatter
    const long EH = E * HEADS;
    edge_scores<<<blocks(EH, BLK), BLK, 0, stream>>>(Qb, Kb, rel, ei, et, scores, segmax, E);
    edge_exp<<<blocks(EH, BLK), BLK, 0, stream>>>(ei, scores, segmax, segsum, E);
    edge_scatter<<<blocks(EH, BLK), BLK, 0, stream>>>(ei, scores, segsum, Vb, agg, E);

    // 4) output projection: agg @ Wo + bo (reuse Xbf buffer for bf16 agg)
    cvt_f32_to_bf16<<<blocks(N * HIDDEN, BLK), BLK, 0, stream>>>(agg, Xbf, N * HIDDEN);
    gemm128_bf16_wmma<<<gemmBlocks, BLK, 0, stream>>>(Xbf, Wot, bo, (float*)d_out, N);
}
